// MPMC_net_8572754722920
// MI455X (gfx1250) — matrix operand order repaired and checked by reading, not compiled, not run on hardware
//
#include <hip/hip_runtime.h>
#include <hip/hip_bf16.h>
#include <math.h>
#include <stdint.h>

typedef __attribute__((ext_vector_type(16))) _Float16 v16h;
typedef __attribute__((ext_vector_type(8)))  _Float16 v8h;
typedef __attribute__((ext_vector_type(8)))  float    v8f;
typedef __attribute__((ext_vector_type(4)))  float    v4f;

#define NBATCH   8
#define NSAMPLES 2048
#define DIMX     4
#define NHID     64
#define NLAYERS  3
#define DEG      16
#define NNODES   (NBATCH * NSAMPLES)   /* 16384 */
#define NEDGES   (NNODES * DEG)        /* 262144 */
#define EPSV     1e-5f

__device__ __forceinline__ v8f wmma16(v16h a, v16h b, v8f c) {
    return __builtin_amdgcn_wmma_f32_16x16x32_f16(
        /*neg_a=*/false, a, /*neg_b=*/false, b,
        /*c_mod=*/(short)0, c, /*reuse_a=*/false, /*reuse_b=*/false);
}

// Async global->LDS copy of 128B (8 x B128), per lane. INST_OFFSET is added to
// BOTH the LDS and global address (CDNA5 ISA), so one LDS base
// VGPR + one 64-bit global base covers the whole row.
__device__ __forceinline__ void async_copy_row128(unsigned ldsOff, const void* gptr) {
    asm volatile("global_load_async_to_lds_b128 %0, %1, off"             :: "v"(ldsOff), "v"(gptr) : "memory");
    asm volatile("global_load_async_to_lds_b128 %0, %1, off offset:16"   :: "v"(ldsOff), "v"(gptr) : "memory");
    asm volatile("global_load_async_to_lds_b128 %0, %1, off offset:32"   :: "v"(ldsOff), "v"(gptr) : "memory");
    asm volatile("global_load_async_to_lds_b128 %0, %1, off offset:48"   :: "v"(ldsOff), "v"(gptr) : "memory");
    asm volatile("global_load_async_to_lds_b128 %0, %1, off offset:64"   :: "v"(ldsOff), "v"(gptr) : "memory");
    asm volatile("global_load_async_to_lds_b128 %0, %1, off offset:80"   :: "v"(ldsOff), "v"(gptr) : "memory");
    asm volatile("global_load_async_to_lds_b128 %0, %1, off offset:96"   :: "v"(ldsOff), "v"(gptr) : "memory");
    asm volatile("global_load_async_to_lds_b128 %0, %1, off offset:112"  :: "v"(ldsOff), "v"(gptr) : "memory");
}

__device__ __forceinline__ void wait_async0() {
    asm volatile("s_wait_asynccnt 0" ::: "memory");
}

// Generic pointer to LDS byte offset (ISA 10.2: the low 32 bits of the address).
__device__ __forceinline__ unsigned lds_off(const void* p) {
    return (unsigned)(uintptr_t)p;
}

// ---------------------------------------------------------------------------
// Weight prep: transpose each layer's f32 weights into f16 column-major so a
// WMMA B fragment is 16 contiguous halves per lane.
// Per-layer layout in wT (halves): m1T[64][128] @0, m2T[64][64] @8192,
//                                  u1T[64][128] @12288, u2T[64][64] @20480
// ---------------------------------------------------------------------------
__global__ void prep_weights_k(const float* __restrict__ m1w,
                               const float* __restrict__ m2w,
                               const float* __restrict__ u1w,
                               const float* __restrict__ u2w,
                               _Float16* __restrict__ wT) {
    int t = threadIdx.x; // 256 threads, 1 block
    for (int l = 0; l < NLAYERS; ++l) {
        _Float16* m1T = wT + l * 24576;
        _Float16* m2T = m1T + 8192;
        _Float16* u1T = m2T + 4096;
        _Float16* u2T = u1T + 8192;
        const float* w;
        w = m1w + l * 128 * 64;
        for (int i = t; i < 128 * 64; i += 256) {
            int k = i >> 6, n = i & 63;
            m1T[n * 128 + k] = (_Float16)w[i];
        }
        w = m2w + l * 64 * 64;
        for (int i = t; i < 64 * 64; i += 256) {
            int k = i >> 6, n = i & 63;
            m2T[n * 64 + k] = (_Float16)w[i];
        }
        w = u1w + l * 128 * 64;
        for (int i = t; i < 128 * 64; i += 256) {
            int k = i >> 6, n = i & 63;
            u1T[n * 128 + k] = (_Float16)w[i];
        }
        w = u2w + l * 64 * 64;
        for (int i = t; i < 64 * 64; i += 256) {
            int k = i >> 6, n = i & 63;
            u2T[n * 64 + k] = (_Float16)w[i];
        }
    }
}

// ---------------------------------------------------------------------------
// Encoder: hH = (f16)(x @ enc_w + enc_b)   ([N,4] @ [4,64])
// Converted to f16 ONCE here; all per-edge gathers are then raw byte copies.
// ---------------------------------------------------------------------------
__global__ __launch_bounds__(256) void encoder_k(const float* __restrict__ x,
                                                 const float* __restrict__ encw,
                                                 const float* __restrict__ encb,
                                                 _Float16* __restrict__ hH) {
    __shared__ float w[4 * 64];
    __shared__ float b[64];
    int t = threadIdx.x;
    w[t] = encw[t];
    if (t < 64) b[t] = encb[t];
    __syncthreads();
    int node = blockIdx.x * 256 + t;
    float x0 = x[node * 4 + 0], x1 = x[node * 4 + 1];
    float x2 = x[node * 4 + 2], x3 = x[node * 4 + 3];
    _Float16* hp = hH + (size_t)node * 64;
    #pragma unroll 8
    for (int j = 0; j < 64; ++j) {
        hp[j] = (_Float16)(b[j] + x0 * w[j] + x1 * w[64 + j] + x2 * w[128 + j] + x3 * w[192 + j]);
    }
}

__global__ void zero_k(float* __restrict__ p, int n) {
    int i = blockIdx.x * blockDim.x + threadIdx.x;
    if (i < n) p[i] = 0.f;
}

// ---------------------------------------------------------------------------
// Fused edge MLP + scatter-add.  Per block: 64 edges, 4 waves.
//   A tile (LDS, f16): [64 edges][ x_i=hH[tgt] (64) | x_j=hH[src] (64) ]
//   gather = per-lane async global->LDS DMA (no VALU, ASYNCcnt tracked)
//   m    = relu(A @ m1 + b1)      (4 K-chunks of 32 -> 16 WMMA/wave)
//   out  = relu(m @ m2 + b2)      (2 K-chunks       ->  8 WMMA/wave)
//   agg[tgt] += out               (fused segment_sum, atomic f32)
// ---------------------------------------------------------------------------
__global__ __launch_bounds__(128) void edge_k(const _Float16* __restrict__ hH,
                                              const long long* __restrict__ ei,
                                              const _Float16* __restrict__ m1T,
                                              const float* __restrict__ m1b,
                                              const _Float16* __restrict__ m2T,
                                              const float* __restrict__ m2b,
                                              float* __restrict__ agg) {
    __shared__ _Float16 Ae[64][128];
    __shared__ _Float16 Me[64][64];
    __shared__ int tg[64];

    const long long* srcIdx = ei;           // edge_index[0]
    const long long* tgtIdx = ei + NEDGES;  // edge_index[1]

    int t = threadIdx.x;
    int eBase = blockIdx.x * 64;

    // ---- gather: 2 threads per edge (part0 = x_i/tgt, part1 = x_j/src),
    //      each copies one 128-byte feature row via async DMA to LDS.
    {
        int el = t >> 1;
        int part = t & 1;
        int e = eBase + el;
        int node = (int)(part ? srcIdx[e] : tgtIdx[e]);
        if (part == 0) tg[el] = (int)tgtIdx[e];
        const _Float16* hp = hH + (size_t)node * 64;
        async_copy_row128(lds_off(&Ae[el][part * 64]), hp);
    }

    int lane   = t & 31;
    int wid    = t >> 5;            // wave -> 16-column slice
    int colL   = lane & 15;         // N within tile (and A row within tile)
    int hiHalf = lane >> 4;         // 0: lanes 0-15, 1: lanes 16-31
    int colG   = wid * 16 + colL;

    // ---- resident B fragments (lane=column, 16 contiguous K, same order as A);
    //      issued while the async gather is in flight.
    v16h B1[4];
    #pragma unroll
    for (int kc = 0; kc < 4; ++kc)
        B1[kc] = *(const v16h*)(m1T + colG * 128 + kc * 32 + hiHalf * 16);
    v16h B2[2];
    #pragma unroll
    for (int kc = 0; kc < 2; ++kc)
        B2[kc] = *(const v16h*)(m2T + colG * 64 + kc * 32 + hiHalf * 16);
    float b1 = m1b[colG];
    float b2 = m2b[colG];

    wait_async0();
    __syncthreads();

    // ---- GEMM1: K=128
    #pragma unroll
    for (int mt = 0; mt < 4; ++mt) {
        v8f c = {};
        int row = mt * 16 + colL;
        #pragma unroll
        for (int kc = 0; kc < 4; ++kc) {
            int kb = kc * 32 + hiHalf * 16;  // A: run {kb..kb+15}
            v8h lo = *(const v8h*)&Ae[row][kb];
            v8h hi = *(const v8h*)&Ae[row][kb + 8];
            v16h a = __builtin_shufflevector(lo, hi, 0, 1, 2, 3, 4, 5, 6, 7,
                                             8, 9, 10, 11, 12, 13, 14, 15);
            c = wmma16(a, B1[kc], c);
        }
        #pragma unroll
        for (int v = 0; v < 8; ++v) {        // C: (M = v + 8*hiHalf, N = colL)
            float r = c[v] + b1;
            r = r > 0.f ? r : 0.f;
            Me[mt * 16 + v + hiHalf * 8][colG] = (_Float16)r;
        }
    }
    __syncthreads();

    // ---- GEMM2: K=64, then scatter-add
    #pragma unroll
    for (int mt = 0; mt < 4; ++mt) {
        v8f c = {};
        int row = mt * 16 + colL;
        #pragma unroll
        for (int kc = 0; kc < 2; ++kc) {
            int kb = kc * 32 + hiHalf * 16;
            v8h lo = *(const v8h*)&Me[row][kb];
            v8h hi = *(const v8h*)&Me[row][kb + 8];
            v16h a = __builtin_shufflevector(lo, hi, 0, 1, 2, 3, 4, 5, 6, 7,
                                             8, 9, 10, 11, 12, 13, 14, 15);
            c = wmma16(a, B2[kc], c);
        }
        #pragma unroll
        for (int v = 0; v < 8; ++v) {
            float r = c[v] + b2;
            r = r > 0.f ? r : 0.f;
            int orow = mt * 16 + v + hiHalf * 8;
            atomicAdd(&agg[(size_t)tg[orow] * 64 + colG], r);
        }
    }
}

// ---------------------------------------------------------------------------
// Fused node update MLP: h_tmp = relu(relu([h|agg] @ u1 + b1) @ u2 + b2)
// part0 (hH, f16) via async DMA; part1 (agg, f32) converted on the fly.
// ---------------------------------------------------------------------------
__global__ __launch_bounds__(128) void node_k(const _Float16* __restrict__ hH,
                                              const float* __restrict__ agg,
                                              const _Float16* __restrict__ u1T,
                                              const float* __restrict__ u1b,
                                              const _Float16* __restrict__ u2T,
                                              const float* __restrict__ u2b,
                                              float* __restrict__ hout) {
    __shared__ _Float16 Ae[64][128];
    __shared__ _Float16 Me[64][64];

    int t = threadIdx.x;
    int nBase = blockIdx.x * 64;

    {
        int nl = t >> 1;
        int part = t & 1;
        if (part == 0) {
            async_copy_row128(lds_off(&Ae[nl][0]), hH + (size_t)(nBase + nl) * 64);
        } else {
            const float* ap = agg + (size_t)(nBase + nl) * 64;
            #pragma unroll
            for (int j = 0; j < 64; j += 4) {
                v4f v = *(const v4f*)(ap + j);
                Ae[nl][64 + j + 0] = (_Float16)v[0];
                Ae[nl][64 + j + 1] = (_Float16)v[1];
                Ae[nl][64 + j + 2] = (_Float16)v[2];
                Ae[nl][64 + j + 3] = (_Float16)v[3];
            }
        }
    }

    int lane   = t & 31;
    int wid    = t >> 5;
    int colL   = lane & 15;
    int hiHalf = lane >> 4;
    int colG   = wid * 16 + colL;

    v16h B1[4];
    #pragma unroll
    for (int kc = 0; kc < 4; ++kc)
        B1[kc] = *(const v16h*)(u1T + colG * 128 + kc * 32 + hiHalf * 16);
    v16h B2[2];
    #pragma unroll
    for (int kc = 0; kc < 2; ++kc)
        B2[kc] = *(const v16h*)(u2T + colG * 64 + kc * 32 + hiHalf * 16);
    float b1 = u1b[colG];
    float b2 = u2b[colG];

    wait_async0();
    __syncthreads();

    #pragma unroll
    for (int mt = 0; mt < 4; ++mt) {
        v8f c = {};
        int row = mt * 16 + colL;
        #pragma unroll
        for (int kc = 0; kc < 4; ++kc) {
            int kb = kc * 32 + hiHalf * 16;
            v8h lo = *(const v8h*)&Ae[row][kb];
            v8h hi = *(const v8h*)&Ae[row][kb + 8];
            v16h a = __builtin_shufflevector(lo, hi, 0, 1, 2, 3, 4, 5, 6, 7,
                                             8, 9, 10, 11, 12, 13, 14, 15);
            c = wmma16(a, B1[kc], c);
        }
        #pragma unroll
        for (int v = 0; v < 8; ++v) {
            float r = c[v] + b1;
            r = r > 0.f ? r : 0.f;
            Me[mt * 16 + v + hiHalf * 8][colG] = (_Float16)r;
        }
    }
    __syncthreads();

    #pragma unroll
    for (int mt = 0; mt < 4; ++mt) {
        v8f c = {};
        int row = mt * 16 + colL;
        #pragma unroll
        for (int kc = 0; kc < 2; ++kc) {
            int kb = kc * 32 + hiHalf * 16;
            v8h lo = *(const v8h*)&Me[row][kb];
            v8h hi = *(const v8h*)&Me[row][kb + 8];
            v16h a = __builtin_shufflevector(lo, hi, 0, 1, 2, 3, 4, 5, 6, 7,
                                             8, 9, 10, 11, 12, 13, 14, 15);
            c = wmma16(a, B2[kc], c);
        }
        #pragma unroll
        for (int v = 0; v < 8; ++v) {
            float r = c[v] + b2;
            r = r > 0.f ? r : 0.f;
            int orow = mt * 16 + v + hiHalf * 8;
            hout[(size_t)(nBase + orow) * 64 + colG] = r;
        }
    }
}

// ---------------------------------------------------------------------------
// InstanceNorm per graph (block per graph): f32 in -> normalized f16 out
// ---------------------------------------------------------------------------
__global__ __launch_bounds__(256) void inorm_k(const float* __restrict__ hin,
                                               _Float16* __restrict__ hH) {
    __shared__ float ssum[4][64];
    __shared__ float ssq[4][64];
    __shared__ float smean[64], sinv[64];
    int g = blockIdx.x;
    int t = threadIdx.x;
    int c = t & 63, rg = t >> 6;
    const float* base = hin + (size_t)g * NSAMPLES * 64;
    float s = 0.f, q = 0.f;
    for (int r = rg; r < NSAMPLES; r += 4) {
        float v = base[r * 64 + c];
        s += v; q += v * v;
    }
    ssum[rg][c] = s; ssq[rg][c] = q;
    __syncthreads();
    if (t < 64) {
        float S = ssum[0][t] + ssum[1][t] + ssum[2][t] + ssum[3][t];
        float Q = ssq[0][t] + ssq[1][t] + ssq[2][t] + ssq[3][t];
        float mean = S * (1.f / NSAMPLES);
        float var = Q * (1.f / NSAMPLES) - mean * mean;
        smean[t] = mean;
        sinv[t] = rsqrtf(var + EPSV);
    }
    __syncthreads();
    _Float16* ob = hH + (size_t)g * NSAMPLES * 64;
    for (int i = t; i < NSAMPLES * 64; i += 256) {
        int cc = i & 63;
        ob[i] = (_Float16)((base[i] - smean[cc]) * sinv[cc]);
    }
}

// ---------------------------------------------------------------------------
// Decoder: X = sigmoid(hH @ dec_w + dec_b)   ([N,64] @ [64,4])
// ---------------------------------------------------------------------------
__global__ __launch_bounds__(256) void decoder_k(const _Float16* __restrict__ hH,
                                                 const float* __restrict__ decw,
                                                 const float* __restrict__ decb,
                                                 float* __restrict__ X) {
    __shared__ float w[64 * 4];
    __shared__ float b[4];
    int t = threadIdx.x;
    w[t] = decw[t];
    if (t < 4) b[t] = decb[t];
    __syncthreads();
    int node = blockIdx.x * 256 + t;
    float a0 = b[0], a1 = b[1], a2 = b[2], a3 = b[3];
    const _Float16* hp = hH + (size_t)node * 64;
    #pragma unroll 8
    for (int k = 0; k < 64; ++k) {
        float hv = (float)hp[k];
        a0 += hv * w[k * 4 + 0];
        a1 += hv * w[k * 4 + 1];
        a2 += hv * w[k * 4 + 2];
        a3 += hv * w[k * 4 + 3];
    }
    float* xp = X + (size_t)node * 4;
    xp[0] = 1.f / (1.f + expf(-a0));
    xp[1] = 1.f / (1.f + expf(-a1));
    xp[2] = 1.f / (1.f + expf(-a2));
    xp[3] = 1.f / (1.f + expf(-a3));
}

// ---------------------------------------------------------------------------
// Warnock L2-star discrepancy pieces
// accum[0..7]  = per-graph sum_i prod_k(1 - x^2)           (t2 raw)
// accum[8..15] = per-graph sum_ij prod_k(1 - max(xi,xj))   (t3 raw)
// ---------------------------------------------------------------------------
__global__ __launch_bounds__(256) void t2_k(const float* __restrict__ X,
                                            float* __restrict__ accum) {
    __shared__ float red[256];
    int g = blockIdx.x, t = threadIdx.x;
    const float* base = X + (size_t)g * NSAMPLES * 4;
    float s = 0.f;
    for (int r = t; r < NSAMPLES; r += 256) {
        float p = 1.f;
        #pragma unroll
        for (int k = 0; k < 4; ++k) {
            float v = base[r * 4 + k];
            p *= (1.f - v * v);
        }
        s += p;
    }
    red[t] = s;
    __syncthreads();
    for (int o = 128; o > 0; o >>= 1) {
        if (t < o) red[t] += red[t + o];
        __syncthreads();
    }
    if (t == 0) accum[g] = red[0];
}

__global__ __launch_bounds__(256) void t3_k(const float* __restrict__ X,
                                            float* __restrict__ accum) {
    __shared__ float xj[256][4];
    __shared__ float red[256];
    int g = blockIdx.x >> 3;   // 8 blocks per graph
    int ib = blockIdx.x & 7;
    int t = threadIdx.x;
    const float* base = X + (size_t)g * NSAMPLES * 4;
    int i = ib * 256 + t;
    float xi0 = base[i * 4 + 0], xi1 = base[i * 4 + 1];
    float xi2 = base[i * 4 + 2], xi3 = base[i * 4 + 3];
    float s = 0.f;
    for (int jt = 0; jt < NSAMPLES; jt += 256) {
        #pragma unroll
        for (int k = 0; k < 4; ++k) xj[t][k] = base[(jt + t) * 4 + k];
        __syncthreads();
        for (int j = 0; j < 256; ++j) {
            float p = (1.f - fmaxf(xi0, xj[j][0]));
            p *= (1.f - fmaxf(xi1, xj[j][1]));
            p *= (1.f - fmaxf(xi2, xj[j][2]));
            p *= (1.f - fmaxf(xi3, xj[j][3]));
            s += p;
        }
        __syncthreads();
    }
    red[t] = s;
    __syncthreads();
    for (int o = 128; o > 0; o >>= 1) {
        if (t < o) red[t] += red[t + o];
        __syncthreads();
    }
    if (t == 0) atomicAdd(&accum[8 + g], red[0]);
}

__global__ void fin_k(const float* __restrict__ accum, float* __restrict__ out) {
    if (threadIdx.x == 0 && blockIdx.x == 0) {
        const float t1 = 1.f / 81.f;  // (1/3)^4
        float acc = 0.f;
        for (int g = 0; g < 8; ++g) {
            float t2 = accum[g] * (2.f / ((float)NSAMPLES * 16.f));
            float t3 = accum[8 + g] / ((float)NSAMPLES * (float)NSAMPLES);
            acc += t1 - t2 + t3;
        }
        out[0] = acc * (1.f / 8.f);
    }
}

// ---------------------------------------------------------------------------
extern "C" void kernel_launch(void* const* d_in, const int* in_sizes, int n_in,
                              void* d_out, int out_size, void* d_ws, size_t ws_size,
                              hipStream_t stream) {
    const float* x        = (const float*)d_in[0];
    const long long* ei   = (const long long*)d_in[1];
    const float* enc_w    = (const float*)d_in[2];
    const float* enc_b    = (const float*)d_in[3];
    const float* m1_w     = (const float*)d_in[4];
    const float* m1_b     = (const float*)d_in[5];
    const float* m2_w     = (const float*)d_in[6];
    const float* m2_b     = (const float*)d_in[7];
    const float* u1_w     = (const float*)d_in[8];
    const float* u1_b     = (const float*)d_in[9];
    const float* u2_w     = (const float*)d_in[10];
    const float* u2_b     = (const float*)d_in[11];
    const float* dec_w    = (const float*)d_in[12];
    const float* dec_b    = (const float*)d_in[13];

    float* out  = (float*)d_out;          // [0]=loss, [1..65536]=X
    float* Xout = out + 1;

    char* ws = (char*)d_ws;
    const size_t F16SZ = (size_t)NNODES * 64 * sizeof(_Float16); // 2 MB
    const size_t F32SZ = (size_t)NNODES * 64 * sizeof(float);    // 4 MB
    _Float16*  hH    = (_Float16*)(ws);
    float*     hT    = (float*)(ws + F16SZ);
    float*     agg   = (float*)(ws + F16SZ + F32SZ);
    _Float16*  wT    = (_Float16*)(ws + F16SZ + 2 * F32SZ);
    float*     accum = (float*)(ws + F16SZ + 2 * F32SZ + ((73728 * 2 + 255) & ~255));

    // 1) weight prep + encoder (h stored as f16 once)
    prep_weights_k<<<1, 256, 0, stream>>>(m1_w, m2_w, u1_w, u2_w, wT);
    encoder_k<<<NNODES / 256, 256, 0, stream>>>(x, enc_w, enc_b, hH);

    // 2) message-passing layers
    for (int l = 0; l < NLAYERS; ++l) {
        const _Float16* m1T = wT + l * 24576;
        const _Float16* m2T = m1T + 8192;
        const _Float16* u1T = m2T + 4096;
        const _Float16* u2T = u1T + 8192;

        zero_k<<<(NNODES * 64) / 256, 256, 0, stream>>>(agg, NNODES * 64);
        edge_k<<<NEDGES / 64, 128, 0, stream>>>(hH, ei, m1T, m1_b + l * 64,
                                                m2T, m2_b + l * 64, agg);
        node_k<<<NNODES / 64, 128, 0, stream>>>(hH, agg, u1T, u1_b + l * 64,
                                                u2T, u2_b + l * 64, hT);
        inorm_k<<<NBATCH, 256, 0, stream>>>(hT, hH);
    }

    // 3) decoder -> X (into d_out), then loss
    decoder_k<<<NNODES / 256, 256, 0, stream>>>(hH, dec_w, dec_b, Xout);
    zero_k<<<1, 256, 0, stream>>>(accum, 16);
    t2_k<<<NBATCH, 256, 0, stream>>>(Xout, accum);
    t3_k<<<NBATCH * 8, 256, 0, stream>>>(Xout, accum);
    fin_k<<<1, 1, 0, stream>>>(accum, out);
}
